// VGG_45526653337976
// MI455X (gfx1250) — matrix operand order repaired and checked by reading, not compiled, not run on hardware
//
#include <hip/hip_runtime.h>
#include <hip/hip_bf16.h>
#include <stdint.h>

// ---------------------------------------------------------------------------
// CDNA5 WMMA (wave32): v_wmma_f32_16x16x32_bf16
// ---------------------------------------------------------------------------
typedef __bf16 v16bf __attribute__((ext_vector_type(16)));
typedef float  v8f   __attribute__((ext_vector_type(8)));

static __device__ __forceinline__ unsigned short f2bf(float f) {
    unsigned u = __builtin_bit_cast(unsigned, f);
    u += 0x7FFFu + ((u >> 16) & 1u);            // RNE
    return (unsigned short)(u >> 16);
}
static __device__ __forceinline__ float bf2f(unsigned short h) {
    return __builtin_bit_cast(float, (unsigned)h << 16);
}

// ---------------------------------------------------------------------------
// Weight packing
//   layer 0  : K-major  [Cout][Kpad=32], k = cin*9 + tap, zero tail
//   layers 1+: tap-major [Cout][9*Cin],  k' = tap*Cin + cin  (Cin % 32 == 0)
// ---------------------------------------------------------------------------
__global__ void pack_w_kmajor(const float* __restrict__ w, unsigned short* __restrict__ wp,
                              int Cout, int K, int Kpad) {
    size_t total = (size_t)Cout * Kpad;
    for (size_t i = (size_t)blockIdx.x * blockDim.x + threadIdx.x; i < total;
         i += (size_t)gridDim.x * blockDim.x) {
        int k  = (int)(i % (size_t)Kpad);
        int co = (int)(i / (size_t)Kpad);
        wp[i] = f2bf((k < K) ? w[(size_t)co * K + k] : 0.0f);
    }
}

__global__ void pack_w_tap(const float* __restrict__ w, unsigned short* __restrict__ wp,
                           int Cout, int Cin) {
    const int Kw = 9 * Cin;
    size_t total = (size_t)Cout * Kw;
    for (size_t i = (size_t)blockIdx.x * blockDim.x + threadIdx.x; i < total;
         i += (size_t)gridDim.x * blockDim.x) {
        int r   = (int)(i % (size_t)Kw);
        int co  = (int)(i / (size_t)Kw);
        int tap = r / Cin;
        int cin = r - tap * Cin;
        wp[i] = f2bf(w[(size_t)co * Kw + (size_t)cin * 9 + tap]);
    }
}

// ---------------------------------------------------------------------------
// Layer 0: Cin=3, Cout=64, f32 NCHW input -> bf16 NHWC output (one 32-K chunk)
// ---------------------------------------------------------------------------
__global__ void __launch_bounds__(256)
conv0_wmma(const float* __restrict__ in, const unsigned short* __restrict__ wp,
           const float* __restrict__ bias, unsigned short* __restrict__ out,
           int H, int W, int nTilesN)
{
    const int Cin = 3, Cout = 64, Kreal = 27;
    const int HW = H * W, Npix = 2 * HW;

    const int wave = blockIdx.x * (blockDim.x >> 5) + (threadIdx.x >> 5);
    const int tM = wave / nTilesN;
    const int tN = wave - tM * nTilesN;
    if (tM >= (Cout >> 4)) return;

    const int lane = threadIdx.x & 31;
    const int hi   = lane & 16;
    const int m0   = tM << 4;
    const int col  = (tN << 4) + (lane & 15);
    const bool colOK = (col < Npix);
    int b = 0, y = 0, x = 0;
    if (colOK) { b = col / HW; int r = col - b * HW; y = r / W; x = r - y * W; }

    const unsigned short* wrow = wp + (size_t)(m0 + (lane & 15)) * 32 + (hi ? 8 : 0);
    union { v16bf v; uint4 q[2]; } ua, ub;
    ua.q[0] = *(const uint4*)(wrow);
    ua.q[1] = *(const uint4*)(wrow + 16);

    #pragma unroll
    for (int j = 0; j < 16; ++j) {
        float v = 0.0f;
        const int kk = (hi >> 1) + (j & 7) + ((j >> 3) << 4);
        if (colOK && kk < Kreal) {
            const int cin = kk / 9;
            const int rem = kk - cin * 9;
            const int iy  = y + rem / 3 - 1;
            const int ix  = x + (rem % 3) - 1;
            if ((unsigned)iy < (unsigned)H && (unsigned)ix < (unsigned)W)
                v = in[(((size_t)b * Cin + cin) * H + iy) * W + ix];
        }
        ((unsigned short*)&ub)[j] = f2bf(v);
    }

    v8f acc = {};
    acc = __builtin_amdgcn_wmma_f32_16x16x32_bf16(false, ua.v, false, ub.v,
                                                  (short)0, acc, false, false);
    if (colOK) {
        const int mofs = hi ? 8 : 0;
        const float4* bp = (const float4*)(bias + m0 + mofs);
        const float4 b0 = bp[0], b1 = bp[1];
        union { uint4 q; unsigned short u[8]; } st;
        st.u[0] = f2bf(fmaxf(acc[0] + b0.x, 0.0f));
        st.u[1] = f2bf(fmaxf(acc[1] + b0.y, 0.0f));
        st.u[2] = f2bf(fmaxf(acc[2] + b0.z, 0.0f));
        st.u[3] = f2bf(fmaxf(acc[3] + b0.w, 0.0f));
        st.u[4] = f2bf(fmaxf(acc[4] + b1.x, 0.0f));
        st.u[5] = f2bf(fmaxf(acc[5] + b1.y, 0.0f));
        st.u[6] = f2bf(fmaxf(acc[6] + b1.z, 0.0f));
        st.u[7] = f2bf(fmaxf(acc[7] + b1.w, 0.0f));
        *(uint4*)(out + ((size_t)(b * H + y) * W + x) * Cout + m0 + mofs) = st.q;
    }
}

// ---------------------------------------------------------------------------
// Fast conv, M-blocked x4: each wave computes a 64(Cout) x 16(pixel) tile.
// bf16 NHWC in/out, tap-major bf16 weights, Cin % 32 == 0, Cout % 64 == 0.
// Per 32-K chunk: 8x b128 (A) + 2x b128 (B, shared) + 4x v_wmma  (10 vmem : 4 wmma)
// ---------------------------------------------------------------------------
__global__ void __launch_bounds__(256)
conv3x3_wmma_nhwc(const unsigned short* __restrict__ in,   // bf16 [2,H,W,Cin]
                  const unsigned short* __restrict__ wp,   // bf16 [Cout, 9*Cin] tap-major
                  const float* __restrict__ bias,
                  unsigned short* __restrict__ out,        // bf16 [2,H,W,Cout]
                  int Cin, int Cout, int H, int W, int nTilesN)
{
    const int HW = H * W, Npix = 2 * HW;
    const int Kw = 9 * Cin;
    const int nMG = Cout >> 6;                 // 64-channel groups

    const int wave = blockIdx.x * (blockDim.x >> 5) + (threadIdx.x >> 5);
    const int tMG = wave / nTilesN;
    const int tN  = wave - tMG * nTilesN;
    if (tMG >= nMG) return;

    const int lane = threadIdx.x & 31;
    const int hi   = lane & 16;                // 0 / 16
    const int m0   = tMG << 6;
    const int col  = (tN << 4) + (lane & 15);
    const bool colOK = (col < Npix);
    int b = 0, y = 0, x = 0;
    if (colOK) { b = col / HW; int r = col - b * HW; y = r / W; x = r - y * W; }

    // four A-row pointers, 16 channels apart
    const size_t rstep = (size_t)16 * Kw;
    const unsigned short* wrow0 = wp + (size_t)(m0 + (lane & 15)) * Kw + (hi ? 8 : 0);

    v8f acc0 = {}, acc1 = {}, acc2 = {}, acc3 = {};
    #pragma unroll
    for (int tap = 0; tap < 9; ++tap) {
        const int dy = tap / 3, dx = tap - dy * 3;          // constants after unroll
        const int iy = y + dy - 1, ix = x + dx - 1;
        const bool ok = colOK && ((unsigned)iy < (unsigned)H) && ((unsigned)ix < (unsigned)W);
        const unsigned short* pB = in + ((size_t)(b * H + iy) * W + ix) * Cin + (hi >> 1);
        const unsigned short* pA = wrow0 + tap * Cin;
        for (int cb = 0; cb < Cin; cb += 32) {
            union { v16bf v; uint4 q[2]; } ub;
            if (ok) {
                ub.q[0] = *(const uint4*)(pB + cb);          // cin cb+hi/2 .. +7
                ub.q[1] = *(const uint4*)(pB + cb + 16);     // cin cb+16+hi/2 .. +7
            } else {
                ub.q[0] = make_uint4(0u, 0u, 0u, 0u);
                ub.q[1] = make_uint4(0u, 0u, 0u, 0u);
            }
            union { v16bf v; uint4 q[2]; } a0, a1, a2, a3;
            a0.q[0] = *(const uint4*)(pA + cb);
            a0.q[1] = *(const uint4*)(pA + cb + 16);
            a1.q[0] = *(const uint4*)(pA + rstep + cb);
            a1.q[1] = *(const uint4*)(pA + rstep + cb + 16);
            a2.q[0] = *(const uint4*)(pA + 2 * rstep + cb);
            a2.q[1] = *(const uint4*)(pA + 2 * rstep + cb + 16);
            a3.q[0] = *(const uint4*)(pA + 3 * rstep + cb);
            a3.q[1] = *(const uint4*)(pA + 3 * rstep + cb + 16);
            acc0 = __builtin_amdgcn_wmma_f32_16x16x32_bf16(false, a0.v, false, ub.v,
                                                           (short)0, acc0, false, false);
            acc1 = __builtin_amdgcn_wmma_f32_16x16x32_bf16(false, a1.v, false, ub.v,
                                                           (short)0, acc1, false, false);
            acc2 = __builtin_amdgcn_wmma_f32_16x16x32_bf16(false, a2.v, false, ub.v,
                                                           (short)0, acc2, false, false);
            acc3 = __builtin_amdgcn_wmma_f32_16x16x32_bf16(false, a3.v, false, ub.v,
                                                           (short)0, acc3, false, false);
        }
    }

    if (colOK) {
        const int mofs = hi ? 8 : 0;
        unsigned short* op = out + ((size_t)(b * H + y) * W + x) * Cout + m0 + mofs;
        v8f accs[4] = {acc0, acc1, acc2, acc3};
        #pragma unroll
        for (int t = 0; t < 4; ++t) {
            const float4* bp = (const float4*)(bias + m0 + t * 16 + mofs);
            const float4 b0 = bp[0], b1 = bp[1];
            union { uint4 q; unsigned short u[8]; } st;
            st.u[0] = f2bf(fmaxf(accs[t][0] + b0.x, 0.0f));
            st.u[1] = f2bf(fmaxf(accs[t][1] + b0.y, 0.0f));
            st.u[2] = f2bf(fmaxf(accs[t][2] + b0.z, 0.0f));
            st.u[3] = f2bf(fmaxf(accs[t][3] + b0.w, 0.0f));
            st.u[4] = f2bf(fmaxf(accs[t][4] + b1.x, 0.0f));
            st.u[5] = f2bf(fmaxf(accs[t][5] + b1.y, 0.0f));
            st.u[6] = f2bf(fmaxf(accs[t][6] + b1.z, 0.0f));
            st.u[7] = f2bf(fmaxf(accs[t][7] + b1.w, 0.0f));
            *(uint4*)(op + t * 16) = st.q;
        }
    }
}

// ---------------------------------------------------------------------------
// 2x2 maxpool on bf16 NHWC: [2,H,W,C] -> [2,H/2,W/2,C]
// ---------------------------------------------------------------------------
__global__ void maxpool2x2_nhwc(const unsigned short* __restrict__ in,
                                unsigned short* __restrict__ out, int C, int H, int W) {
    const int Ho = H >> 1, Wo = W >> 1;
    const size_t total = (size_t)2 * Ho * Wo * C;
    for (size_t i = (size_t)blockIdx.x * blockDim.x + threadIdx.x; i < total;
         i += (size_t)gridDim.x * blockDim.x) {
        int c  = (int)(i % (size_t)C); size_t t = i / (size_t)C;
        int xo = (int)(t % (size_t)Wo); t /= (size_t)Wo;
        int yo = (int)(t % (size_t)Ho);
        int b  = (int)(t / (size_t)Ho);
        const unsigned short* p = in + ((size_t)(b * H + 2 * yo) * W + 2 * xo) * C + c;
        const size_t rs = (size_t)W * C;
        float v = fmaxf(fmaxf(bf2f(p[0]), bf2f(p[C])),
                        fmaxf(bf2f(p[rs]), bf2f(p[rs + C])));
        out[i] = f2bf(v);
    }
}

// ---------------------------------------------------------------------------
// Per-channel min/max over batch+spatial on bf16 NHWC (block per channel)
// ---------------------------------------------------------------------------
__global__ void chan_minmax_nhwc(const unsigned short* __restrict__ x,
                                 float* __restrict__ mn, float* __restrict__ mx,
                                 int C, int NP /* 2*H*W */) {
    __shared__ float smn[256];
    __shared__ float smx[256];
    const int c = blockIdx.x;
    float lmn = 3.4e38f, lmx = -3.4e38f;
    for (int p = threadIdx.x; p < NP; p += blockDim.x) {
        const float v = bf2f(x[(size_t)p * C + c]);
        lmn = fminf(lmn, v);
        lmx = fmaxf(lmx, v);
    }
    smn[threadIdx.x] = lmn; smx[threadIdx.x] = lmx;
    __syncthreads();
    for (int s = 128; s > 0; s >>= 1) {
        if (threadIdx.x < (unsigned)s) {
            smn[threadIdx.x] = fminf(smn[threadIdx.x], smn[threadIdx.x + s]);
            smx[threadIdx.x] = fmaxf(smx[threadIdx.x], smx[threadIdx.x + s]);
        }
        __syncthreads();
    }
    if (threadIdx.x == 0) { mn[c] = smn[0]; mx[c] = smx[0]; }
}

__global__ void quant_apply_nhwc(unsigned short* __restrict__ x,
                                 const float* __restrict__ mn, const float* __restrict__ mx,
                                 int C, size_t total) {
    for (size_t i = (size_t)blockIdx.x * blockDim.x + threadIdx.x; i < total;
         i += (size_t)gridDim.x * blockDim.x) {
        const int c = (int)(i % (size_t)C);
        const float lo = mn[c];
        const float zd = mx[c] - lo;
        float v = bf2f(x[i]);
        if (zd > 0.0f)
            v = lo + rintf((v - lo) * (255.0f / zd)) * (zd * (1.0f / 255.0f));
        x[i] = f2bf(v);
    }
}

// bf16 NHWC [2,7,7,512] -> f32 [2, 25088] in NCHW-flatten order (c*49 + y*7 + x)
__global__ void repack_flat(const unsigned short* __restrict__ act,
                            float* __restrict__ flat) {
    const int i = blockIdx.x * blockDim.x + threadIdx.x;
    if (i >= 2 * 25088) return;
    const int b = i / 25088;
    const int k = i - b * 25088;
    const int c = k / 49;
    const int s = k - c * 49;                 // y*7 + x
    flat[i] = bf2f(act[(size_t)b * 25088 + (size_t)s * 512 + c]);
}

// ---------------------------------------------------------------------------
// FC (N=2, bandwidth-bound): K-split streaming partials + f32 atomics.
// ---------------------------------------------------------------------------
__global__ void fc_partial(const float* __restrict__ in, const float* __restrict__ w,
                           float* __restrict__ acc, int din, int dout, int kchunk) {
    const int o = blockIdx.x * blockDim.x + threadIdx.x;
    if (o >= dout) return;
    const int k0 = blockIdx.y * kchunk;
    int k1 = k0 + kchunk; if (k1 > din) k1 = din;
    const float* in1 = in + din;
    float s0 = 0.0f, s1 = 0.0f;
    for (int k = k0; k < k1; ++k) {
        __builtin_prefetch(&w[(size_t)(k + 16) * dout + o], 0, 1); // global_prefetch_b8
        const float wv = w[(size_t)k * dout + o];
        s0 = fmaf(in[k],  wv, s0);
        s1 = fmaf(in1[k], wv, s1);
    }
    atomicAdd(&acc[o], s0);
    atomicAdd(&acc[(size_t)dout + o], s1);
}

__global__ void fc_finish(const float* __restrict__ acc, const float* __restrict__ bias,
                          float* __restrict__ out, int dout, int relu) {
    const int i = blockIdx.x * blockDim.x + threadIdx.x;
    if (i >= 2 * dout) return;
    const int o = i % dout;
    float v = acc[i] + bias[o];
    if (relu) v = fmaxf(v, 0.0f);
    out[i] = v;
}

// ---------------------------------------------------------------------------
// Host orchestration
// ---------------------------------------------------------------------------
extern "C" void kernel_launch(void* const* d_in, const int* in_sizes, int n_in,
                              void* d_out, int out_size, void* d_ws, size_t ws_size,
                              hipStream_t stream) {
    (void)in_sizes; (void)n_in; (void)out_size; (void)ws_size;

    const float* x = (const float*)d_in[0];
    const float* cw[13]; const float* cb[13];
    for (int i = 0; i < 13; ++i) {
        cw[i] = (const float*)d_in[1 + 2 * i];
        cb[i] = (const float*)d_in[2 + 2 * i];
    }
    const float* fw1 = (const float*)d_in[27]; const float* fb1 = (const float*)d_in[28];
    const float* fw2 = (const float*)d_in[29]; const float* fb2 = (const float*)d_in[30];
    const float* fw3 = (const float*)d_in[31]; const float* fb3 = (const float*)d_in[32];

    static const int CIN [13] = {3,64,64,128,128,256,256,256,512,512,512,512,512};
    static const int COUT[13] = {64,64,128,128,256,256,256,512,512,512,512,512,512};
    static const int POOL[13] = {0,1,0,1,0,0,1,0,0,1,0,0,1};

    // ---- carve workspace ----
    char* wsb = (char*)d_ws;
    size_t off = 0;
    auto carve = [&](size_t bytes) -> void* {
        void* p = wsb + off;
        off += (bytes + 255) & ~(size_t)255;
        return p;
    };
    const size_t actElems = (size_t)2 * 224 * 224 * 64;          // max activation
    unsigned short* actA = (unsigned short*)carve(actElems * 2); // 12.9 MB
    unsigned short* actB = (unsigned short*)carve(actElems * 2);

    size_t wOff[13], totW = 0;
    wOff[0] = 0; totW += (size_t)64 * 32;                        // layer0 K-major padded
    for (int i = 1; i < 13; ++i) { wOff[i] = totW; totW += (size_t)COUT[i] * 9 * CIN[i]; }
    unsigned short* wpk = (unsigned short*)carve(totW * 2);      // ~29.4 MB

    float* mnb   = (float*)carve(64 * sizeof(float));
    float* mxb   = (float*)carve(64 * sizeof(float));
    float* flat  = (float*)carve((size_t)2 * 25088 * sizeof(float));
    float* fcacc = (float*)carve((size_t)2 * 4096 * sizeof(float));
    float* fcb0  = (float*)carve((size_t)2 * 4096 * sizeof(float));
    float* fcb1  = (float*)carve((size_t)2 * 4096 * sizeof(float));

    // ---- pack weights ----
    pack_w_kmajor<<<(64 * 32 + 255) / 256, 256, 0, stream>>>(cw[0], wpk, 64, 27, 32);
    for (int i = 1; i < 13; ++i) {
        const size_t elems = (size_t)COUT[i] * 9 * CIN[i];
        pack_w_tap<<<(int)((elems + 255) / 256), 256, 0, stream>>>(cw[i], wpk + wOff[i],
                                                                   COUT[i], CIN[i]);
    }

    // ---- conv chain (bf16 NHWC activations) ----
    int H = 224, W = 224;
    unsigned short* buf[2] = {actA, actB};
    int cur = 1;

    // layer 0: f32 NCHW input -> bf16 NHWC
    {
        const int nTilesN = (2 * H * W + 15) / 16;
        const int waves = (64 / 16) * nTilesN;
        conv0_wmma<<<(waves + 7) / 8, 256, 0, stream>>>(x, wpk, cb[0], actA, H, W, nTilesN);
    }
    unsigned short* src = actA;

    for (int i = 1; i < 13; ++i) {
        const int Cin = CIN[i], Cout = COUT[i];
        unsigned short* dst = buf[cur];
        const int nTilesN = (2 * H * W + 15) / 16;
        const int waves = (Cout / 64) * nTilesN;          // M-blocked x4
        conv3x3_wmma_nhwc<<<(waves + 7) / 8, 256, 0, stream>>>(src, wpk + wOff[i], cb[i],
                                                               dst, Cin, Cout, H, W, nTilesN);
        src = dst; cur ^= 1;

        if (POOL[i]) {
            unsigned short* pdst = buf[cur];
            const size_t tot = (size_t)2 * Cout * (H / 2) * (W / 2);
            maxpool2x2_nhwc<<<(int)((tot + 255) / 256), 256, 0, stream>>>(src, pdst, Cout, H, W);
            H >>= 1; W >>= 1;
            src = pdst; cur ^= 1;
        }
        if (i == 1) {   // fake-quant after conv1's ReLU+pool: [2,112,112,64]
            chan_minmax_nhwc<<<64, 256, 0, stream>>>(src, mnb, mxb, 64, 2 * H * W);
            const size_t tot = (size_t)2 * H * W * 64;
            quant_apply_nhwc<<<(int)((tot + 255) / 256), 256, 0, stream>>>(src, mnb, mxb,
                                                                           64, tot);
        }
    }

    // ---- flatten (NCHW order) + FC head ----
    repack_flat<<<(2 * 25088 + 255) / 256, 256, 0, stream>>>(src, flat);

    auto run_fc = [&](const float* inp, const float* w, const float* bias, float* outp,
                      int din, int dout, bool relu, int ksplit) {
        hipMemsetAsync(fcacc, 0, (size_t)2 * dout * sizeof(float), stream);
        const int kchunk = (din + ksplit - 1) / ksplit;
        dim3 grid((dout + 255) / 256, ksplit);
        fc_partial<<<grid, 256, 0, stream>>>(inp, w, fcacc, din, dout, kchunk);
        fc_finish<<<(2 * dout + 255) / 256, 256, 0, stream>>>(fcacc, bias, outp, dout,
                                                              relu ? 1 : 0);
    };
    run_fc(flat, fw1, fb1, fcb0,          25088, 4096, true, 32);
    run_fc(fcb0, fw2, fb2, fcb1,           4096, 4096, true,  8);
    run_fc(fcb1, fw3, fb3, (float*)d_out,  4096,  196, false, 8);
}
